// Moondream3Attention_17978733101589
// MI455X (gfx1250) — matrix operand order repaired and checked by reading, not compiled, not run on hardware
//
#include <hip/hip_runtime.h>

// ---------------------------------------------------------------------------
// Moondream3 attention block on MI455X (gfx1250), bf16 WMMA pipeline.
//
//  1) f32->bf16 conversion of hidden, Wqkv, Wout into workspace
//  2) QKV GEMM: qkv = hidden @ Wqkv^T + bqkv   (WMMA, async-LDS double buffer)
//  3) tau kernel: tok_q/tok_v = tanh(gelu(qkv) @ tau_w^T) + tau_pos (f32 VALU)
//  4) rope/scale kernel: q_bf16[h][t][d] (incl tau_q, d^-1/2), k_bf16, vT_bf16
//  5) causal flash attention per head, WMMA QK^T + PV, online softmax
//  6) out GEMM: out = attn @ Wout^T + bout     (WMMA, async-LDS double buffer)
// ---------------------------------------------------------------------------

typedef unsigned short u16;
typedef __attribute__((ext_vector_type(16))) __bf16 bf16x16;
typedef __attribute__((ext_vector_type(8)))  float  f32x8;
typedef __attribute__((__vector_size__(16))) int    v4i;

#define T_SEQ   2048
#define DIM_    2048
#define NH_     32
#define HD_     64
#define QKV_    6144
#define ROT_    32

// gfx1250 async global->LDS copies (ASYNCcnt); guarded so both toolchains and
// the host pass compile. Fallback: regular load + ds_store staging.
#if defined(__has_builtin)
#  if __has_builtin(__builtin_amdgcn_global_load_async_to_lds_b128) && \
      __has_builtin(__builtin_amdgcn_s_wait_asynccnt)
#    define HAVE_ASYNC_LDS 1
#  endif
#endif

// workspace layout (bytes)
#define OFF_HSB    ((size_t)0)                         // T*DIM bf16       8 MB
#define OFF_WQKVB  (OFF_HSB   + (size_t)T_SEQ*DIM_*2)  // QKV*DIM bf16    25 MB
#define OFF_WOUTB  (OFF_WQKVB + (size_t)QKV_*DIM_*2)   // DIM*DIM bf16     8 MB
#define OFF_QKV    (OFF_WOUTB + (size_t)DIM_*DIM_*2)   // T*QKV f32       50 MB
#define OFF_TAUQ   (OFF_QKV   + (size_t)T_SEQ*QKV_*4)  // T*32 f32
#define OFF_TAUV   (OFF_TAUQ  + (size_t)T_SEQ*NH_*4)
#define OFF_QB     (OFF_TAUV  + (size_t)T_SEQ*NH_*4)   // H*T*D bf16       8 MB
#define OFF_KB     (OFF_QB    + (size_t)T_SEQ*DIM_*2)
#define OFF_VTB    (OFF_KB    + (size_t)T_SEQ*DIM_*2)  // transposed V
#define OFF_ATTNB  (OFF_VTB   + (size_t)T_SEQ*DIM_*2)  // T*DIM bf16

__device__ __forceinline__ u16 f2bf(float f) {
  unsigned u = __float_as_uint(f);
  u += 0x7FFFu + ((u >> 16) & 1u);   // round-to-nearest-even
  return (u16)(u >> 16);
}

union V16 { bf16x16 v; uint4 q[2]; };

// ---------------------------------------------------------------------------
// 1) elementwise f32 -> bf16
// ---------------------------------------------------------------------------
__global__ void cvt_bf16_kernel(const float* __restrict__ src,
                                u16* __restrict__ dst, size_t n) {
  size_t i = (size_t)blockIdx.x * blockDim.x + threadIdx.x;
  size_t stride = (size_t)gridDim.x * blockDim.x;
  for (; i < n; i += stride) dst[i] = f2bf(src[i]);
}

// ---------------------------------------------------------------------------
// 2,6) C[M][N] = A_bf16[M][K] @ W_bf16[N][K]^T + bias[N], f32 out.
// Block = 256 threads (8 waves) -> 128x64 tile; K stepped by 32 with
// double-buffered LDS staging via async global->LDS copies.
// ---------------------------------------------------------------------------
#define KSTEP 32
#define MBLK  128
#define NBLK  64

__device__ __forceinline__ void stage_chunk(const u16* __restrict__ g,
                                            u16* __restrict__ l) {
#if defined(HAVE_ASYNC_LDS)
  __builtin_amdgcn_global_load_async_to_lds_b128(
      (__attribute__((address_space(1))) v4i*)(void*)g,
      (__attribute__((address_space(3))) v4i*)l, 0, 0);
#else
  *(uint4*)l = *(const uint4*)g;
#endif
}

__device__ __forceinline__ void stage_tiles(
    const u16* __restrict__ A, const u16* __restrict__ W,
    u16 (* __restrict__ As)[KSTEP], u16 (* __restrict__ Bs)[KSTEP],
    int m0blk, int n0, int K, int k, int tid)
{
  // A tile: 128 rows x 32 cols bf16 = 512 x 16B chunks, 2 per thread
  #pragma unroll
  for (int i = 0; i < 2; ++i) {
    int c = tid + i * 256;
    int row = c >> 2, q = c & 3;
    stage_chunk(A + (size_t)(m0blk + row) * K + k + q * 8, &As[row][q * 8]);
  }
  // B tile: 64 rows x 32 cols = 256 chunks, 1 per thread
  {
    int row = tid >> 2, q = tid & 3;
    stage_chunk(W + (size_t)(n0 + row) * K + k + q * 8, &Bs[row][q * 8]);
  }
}

__device__ __forceinline__ void stage_join() {
#if defined(HAVE_ASYNC_LDS)
  __builtin_amdgcn_s_wait_asynccnt(0);   // own async copies landed in LDS
#endif
  __syncthreads();                       // all waves' copies visible
}

__global__ __launch_bounds__(256) void gemm_bf16_bias_kernel(
    const u16* __restrict__ A, const u16* __restrict__ W,
    const float* __restrict__ bias, float* __restrict__ C,
    int M, int N, int K)
{
  __shared__ __align__(16) u16 As[2][MBLK][KSTEP];
  __shared__ __align__(16) u16 Bs[2][NBLK][KSTEP];

  const int tid  = threadIdx.x;
  const int lane = tid & 31;
  const int wave = tid >> 5;
  const int hi = lane >> 4;     // half-wave select
  const int ln = lane & 15;
  const int m0blk = blockIdx.y * MBLK;
  const int n0 = blockIdx.x * NBLK;

  f32x8 acc[4];
  #pragma unroll
  for (int j = 0; j < 4; ++j) {
    float b = bias[n0 + j * 16 + ln];
    #pragma unroll
    for (int r = 0; r < 8; ++r) acc[j][r] = b;
  }

  stage_tiles(A, W, As[0], Bs[0], m0blk, n0, K, 0, tid);

  const int ksteps = K / KSTEP;
  for (int kt = 0; kt < ksteps; ++kt) {
    const int buf = kt & 1;
    stage_join();   // tile[buf] ready; all waves done reading tile[buf^1]
    if (kt + 1 < ksteps)
      stage_tiles(A, W, As[buf ^ 1], Bs[buf ^ 1], m0blk, n0, K,
                  (kt + 1) * KSTEP, tid);

    // compute from LDS. Issue all 10 ds_load_b128 up front into distinct
    // registers so the dscnt waits stage down instead of draining to 0
    // before every WMMA.
    V16 a, b[4];
    a.q[0] = *(const uint4*)&As[buf][wave * 16 + ln][8 * hi];
    a.q[1] = *(const uint4*)&As[buf][wave * 16 + ln][16 + 8 * hi];
    #pragma unroll
    for (int j = 0; j < 4; ++j) {
      b[j].q[0] = *(const uint4*)&Bs[buf][j * 16 + ln][8 * hi];
      b[j].q[1] = *(const uint4*)&Bs[buf][j * 16 + ln][16 + 8 * hi];
    }
    #pragma unroll
    for (int j = 0; j < 4; ++j)
      acc[j] = __builtin_amdgcn_wmma_f32_16x16x32_bf16(
          false, a.v, false, b[j].v, (short)0, acc[j], false, false);
  }

  #pragma unroll
  for (int j = 0; j < 4; ++j)
    #pragma unroll
    for (int r = 0; r < 8; ++r)
      C[(size_t)(m0blk + wave * 16 + r + 8 * hi) * N + n0 + j * 16 + ln] =
          acc[j][r];
}

// ---------------------------------------------------------------------------
// 3) tau kernel: tok_q/tok_v = tanh(gelu(qkv_row) . tau_w[h]) + tau_pos
// ---------------------------------------------------------------------------
__global__ __launch_bounds__(256) void tau_kernel(
    const int* __restrict__ positions, const float* __restrict__ qkv,
    const float* __restrict__ tau_alpha,
    const float* __restrict__ tau_wq, const float* __restrict__ tau_wv,
    float* __restrict__ tauq, float* __restrict__ tauv)
{
  __shared__ float feat[2][QKV_];
  const int tbase = blockIdx.x * 2;
  for (int i = threadIdx.x; i < 2 * QKV_; i += blockDim.x) {
    int tl = i / QKV_, k = i - tl * QKV_;
    float x = qkv[(size_t)(tbase + tl) * QKV_ + k];
    feat[tl][k] = 0.5f * x * (1.0f + erff(x * 0.70710678118654752f)); // exact gelu
  }
  __syncthreads();

  int idx = threadIdx.x;            // 128 dots: (2 rows) x (64 outputs)
  if (idx < 128) {
    int o = idx & 63, tl = idx >> 6;
    const float* w = (o < 32) ? (tau_wq + (size_t)o * QKV_)
                              : (tau_wv + (size_t)(o - 32) * QKV_);
    float acc = 0.0f;
    for (int k = 0; k < QKV_; k += 4) {
      float4 wv = *(const float4*)(w + k);
      acc += feat[tl][k] * wv.x + feat[tl][k + 1] * wv.y +
             feat[tl][k + 2] * wv.z + feat[tl][k + 3] * wv.w;
    }
    int t = tbase + tl;
    float pl = logf(fmaxf((float)positions[t] + 1.0f, 1e-6f));
    int hh = o & 31;
    float tp = 0.5f + 1.0f / (1.0f + __expf(-tau_alpha[hh] * pl));
    float val = tanhf(acc) + tp;
    if (o < 32) tauq[t * NH_ + hh] = val;
    else        tauv[t * NH_ + hh] = val;
  }
}

// ---------------------------------------------------------------------------
// 4) rope + tau scaling; q_bf16[h][t][d] (x tau_q x 0.125), k_bf16, vT_bf16
// ---------------------------------------------------------------------------
__global__ __launch_bounds__(256) void rope_scale_kernel(
    const int* __restrict__ positions, const float* __restrict__ qkv,
    const float* __restrict__ tauq, const float* __restrict__ tauv,
    u16* __restrict__ qb, u16* __restrict__ kb, u16* __restrict__ vtb)
{
  const int t = blockIdx.x;
  const float pos = (float)positions[t];
  const float* row = qkv + (size_t)t * QKV_;
  for (int idx = threadIdx.x; idx < DIM_; idx += blockDim.x) {
    int hh = idx >> 6, d = idx & 63;
    float tq = tauq[t * NH_ + hh], tv = tauv[t * NH_ + hh];
    float qv = row[idx], kv = row[DIM_ + idx], vv = row[2 * DIM_ + idx];
    float qo = qv, ko = kv;
    if (d < ROT_) {
      int j = d & 15;
      float inv_freq = __expf(-((float)(2 * j) * (1.0f / 32.0f)) * logf(1.0e6f));
      float ang = pos * inv_freq;
      float s, c;
      __sincosf(ang, &s, &c);
      if (d < 16) {
        float q2 = row[idx + 16], k2 = row[DIM_ + idx + 16];
        qo = qv * c - q2 * s;  ko = kv * c - k2 * s;
      } else {
        float q1 = row[idx - 16], k1 = row[DIM_ + idx - 16];
        qo = qv * c + q1 * s;  ko = kv * c + k1 * s;
      }
    }
    qo *= tq * 0.125f;   // fold tau_q and 1/sqrt(64) into q
    float vo = vv * tv;
    size_t qi = ((size_t)hh * T_SEQ + t) * HD_ + d;
    qb[qi] = f2bf(qo);
    kb[qi] = f2bf(ko);
    vtb[((size_t)(hh * HD_ + d)) * T_SEQ + t] = f2bf(vo);
  }
}

// ---------------------------------------------------------------------------
// 5) causal flash attention: one wave = 16 query rows of one head.
//    Block = 4 waves (64 rows), grid = (T/64, H).
// ---------------------------------------------------------------------------
__global__ __launch_bounds__(128) void attn_kernel(
    const u16* __restrict__ qb, const u16* __restrict__ kb,
    const u16* __restrict__ vtb, u16* __restrict__ attnb)
{
  __shared__ __align__(16) u16 pbuf[4][16][32];   // per-wave P tile (bf16)
  const int lane = threadIdx.x & 31;
  const int wave = threadIdx.x >> 5;
  const int hi = lane >> 4;
  const int ln = lane & 15;
  const int h = blockIdx.y;
  const int t0 = blockIdx.x * 64 + wave * 16;

  // Q A-tiles (M=16 x K=32, two d-chunks), preloaded once
  const u16* qrow = qb + ((size_t)h * T_SEQ + (t0 + ln)) * HD_;
  V16 qa0, qa1;
  qa0.q[0] = *(const uint4*)(qrow + 8 * hi);
  qa0.q[1] = *(const uint4*)(qrow + 16 + 8 * hi);
  qa1.q[0] = *(const uint4*)(qrow + 32 + 8 * hi);
  qa1.q[1] = *(const uint4*)(qrow + 48 + 8 * hi);

  f32x8 O[4];
  float mrow[8], lrow[8];
  #pragma unroll
  for (int r = 0; r < 8; ++r) { mrow[r] = -1.0e30f; lrow[r] = 0.0f; }
  #pragma unroll
  for (int j = 0; j < 4; ++j)
    #pragma unroll
    for (int r = 0; r < 8; ++r) O[j][r] = 0.0f;

  const int stiles = (t0 + 16 + 31) >> 5;   // causal: keys <= t0+15
  for (int st = 0; st < stiles; ++st) {
    const int sb = st * 32;

    // --- scores: S0 = keys sb..sb+15, S1 = keys sb+16..sb+31 ---
    // load all 4 B tiles (2 key groups x 2 d-chunks) before the WMMAs
    f32x8 S0, S1;
    #pragma unroll
    for (int r = 0; r < 8; ++r) { S0[r] = 0.0f; S1[r] = 0.0f; }
    {
      const u16* krow0 = kb + ((size_t)h * T_SEQ + sb + ln) * HD_;
      const u16* krow1 = kb + ((size_t)h * T_SEQ + sb + 16 + ln) * HD_;
      V16 b00, b01, b10, b11;
      b00.q[0] = *(const uint4*)(krow0 + 8 * hi);
      b00.q[1] = *(const uint4*)(krow0 + 16 + 8 * hi);
      b01.q[0] = *(const uint4*)(krow0 + 32 + 8 * hi);
      b01.q[1] = *(const uint4*)(krow0 + 48 + 8 * hi);
      b10.q[0] = *(const uint4*)(krow1 + 8 * hi);
      b10.q[1] = *(const uint4*)(krow1 + 16 + 8 * hi);
      b11.q[0] = *(const uint4*)(krow1 + 32 + 8 * hi);
      b11.q[1] = *(const uint4*)(krow1 + 48 + 8 * hi);
      S0 = __builtin_amdgcn_wmma_f32_16x16x32_bf16(false, qa0.v, false, b00.v, (short)0, S0, false, false);
      S0 = __builtin_amdgcn_wmma_f32_16x16x32_bf16(false, qa1.v, false, b01.v, (short)0, S0, false, false);
      S1 = __builtin_amdgcn_wmma_f32_16x16x32_bf16(false, qa0.v, false, b10.v, (short)0, S1, false, false);
      S1 = __builtin_amdgcn_wmma_f32_16x16x32_bf16(false, qa1.v, false, b11.v, (short)0, S1, false, false);
    }

    // --- causal mask + online softmax (C layout: row = r+8*hi, col = ln) ---
    const int s0c = sb + ln, s1c = sb + 16 + ln;
    #pragma unroll
    for (int r = 0; r < 8; ++r) {
      const int tr = t0 + r + 8 * hi;
      float v0 = (s0c > tr) ? -1.0e30f : S0[r];
      float v1 = (s1c > tr) ? -1.0e30f : S1[r];
      float mx = fmaxf(v0, v1);
      mx = fmaxf(mx, __shfl_xor(mx, 1));
      mx = fmaxf(mx, __shfl_xor(mx, 2));
      mx = fmaxf(mx, __shfl_xor(mx, 4));
      mx = fmaxf(mx, __shfl_xor(mx, 8));
      float mnew = fmaxf(mrow[r], mx);
      float alpha = __expf(mrow[r] - mnew);
      float p0 = __expf(v0 - mnew);
      float p1 = __expf(v1 - mnew);
      float rs = p0 + p1;
      rs += __shfl_xor(rs, 1);
      rs += __shfl_xor(rs, 2);
      rs += __shfl_xor(rs, 4);
      rs += __shfl_xor(rs, 8);
      lrow[r] = lrow[r] * alpha + rs;
      mrow[r] = mnew;
      #pragma unroll
      for (int j = 0; j < 4; ++j) O[j][r] *= alpha;
      pbuf[wave][r + 8 * hi][ln]      = f2bf(p0);
      pbuf[wave][r + 8 * hi][16 + ln] = f2bf(p1);
    }

    // per-wave LDS ordering only (no block barrier: waves diverge in trip count)
    asm volatile("s_wait_dscnt 0" ::: "memory");

    // --- re-layout P into A-matrix layout and do PV ---
    // load P tile + all 4 V B-tiles before the WMMAs
    V16 aP, bv[4];
    aP.q[0] = *(const uint4*)(&pbuf[wave][ln][8 * hi]);
    aP.q[1] = *(const uint4*)(&pbuf[wave][ln][16 + 8 * hi]);
    #pragma unroll
    for (int j = 0; j < 4; ++j) {
      const u16* vrow = vtb + ((size_t)(h * HD_ + j * 16 + ln)) * T_SEQ + sb;
      bv[j].q[0] = *(const uint4*)(vrow + 8 * hi);
      bv[j].q[1] = *(const uint4*)(vrow + 16 + 8 * hi);
    }
    #pragma unroll
    for (int j = 0; j < 4; ++j)
      O[j] = __builtin_amdgcn_wmma_f32_16x16x32_bf16(
          false, aP.v, false, bv[j].v, (short)0, O[j], false, false);
  }

  // --- normalize and emit attn output (bf16, [t][h*64+d]) ---
  float invl[8];
  #pragma unroll
  for (int r = 0; r < 8; ++r) invl[r] = 1.0f / lrow[r];
  #pragma unroll
  for (int j = 0; j < 4; ++j)
    #pragma unroll
    for (int r = 0; r < 8; ++r)
      attnb[(size_t)(t0 + r + 8 * hi) * DIM_ + h * HD_ + j * 16 + ln] =
          f2bf(O[j][r] * invl[r]);
}

// ---------------------------------------------------------------------------
// host-side launcher
// ---------------------------------------------------------------------------
extern "C" void kernel_launch(void* const* d_in, const int* in_sizes, int n_in,
                              void* d_out, int out_size, void* d_ws, size_t ws_size,
                              hipStream_t stream) {
  (void)in_sizes; (void)n_in; (void)out_size; (void)ws_size;
  const int*   positions = (const int*)  d_in[0];
  const float* hidden    = (const float*)d_in[1];
  const float* Wqkv      = (const float*)d_in[2];
  const float* bqkv      = (const float*)d_in[3];
  const float* Wout      = (const float*)d_in[4];
  const float* bout      = (const float*)d_in[5];
  const float* tau_alpha = (const float*)d_in[6];
  const float* tau_wq    = (const float*)d_in[7];
  const float* tau_wv    = (const float*)d_in[8];
  float* out = (float*)d_out;

  char* ws = (char*)d_ws;
  u16*   hsb   = (u16*)  (ws + OFF_HSB);
  u16*   wqkvb = (u16*)  (ws + OFF_WQKVB);
  u16*   woutb = (u16*)  (ws + OFF_WOUTB);
  float* qkv   = (float*)(ws + OFF_QKV);
  float* tauq  = (float*)(ws + OFF_TAUQ);
  float* tauv  = (float*)(ws + OFF_TAUV);
  u16*   qb    = (u16*)  (ws + OFF_QB);
  u16*   kb    = (u16*)  (ws + OFF_KB);
  u16*   vtb   = (u16*)  (ws + OFF_VTB);
  u16*   attnb = (u16*)  (ws + OFF_ATTNB);

  // 1) conversions
  cvt_bf16_kernel<<<4096, 256, 0, stream>>>(hidden, hsb, (size_t)T_SEQ * DIM_);
  cvt_bf16_kernel<<<8192, 256, 0, stream>>>(Wqkv, wqkvb, (size_t)QKV_ * DIM_);
  cvt_bf16_kernel<<<4096, 256, 0, stream>>>(Wout, woutb, (size_t)DIM_ * DIM_);

  // 2) QKV GEMM: [2048,2048] @ [2048,6144] + bias
  gemm_bf16_bias_kernel<<<dim3(QKV_ / 64, T_SEQ / 128), 256, 0, stream>>>(
      hsb, wqkvb, bqkv, qkv, T_SEQ, QKV_, DIM_);

  // 3) tau gates
  tau_kernel<<<T_SEQ / 2, 256, 0, stream>>>(positions, qkv, tau_alpha,
                                            tau_wq, tau_wv, tauq, tauv);

  // 4) rope + scaling + head-major/bf16 repack
  rope_scale_kernel<<<T_SEQ, 256, 0, stream>>>(positions, qkv, tauq, tauv,
                                               qb, kb, vtb);

  // 5) causal flash attention
  attn_kernel<<<dim3(T_SEQ / 64, NH_), 128, 0, stream>>>(qb, kb, vtb, attnb);

  // 6) output projection: [2048,2048] @ [2048,2048] + bias -> d_out (f32)
  gemm_bf16_bias_kernel<<<dim3(DIM_ / 64, T_SEQ / 128), 256, 0, stream>>>(
      attnb, woutb, bout, out, T_SEQ, DIM_, DIM_);
}
